// TUBA_35046933135406
// MI455X (gfx1250) — compile-verified, matrix operand order checked by reading, not run on hardware
//
#include <hip/hip_runtime.h>
#include <hip/hip_bf16.h>

// Problem constants (from reference)
#define NS   1024   // number of samples
#define XD   64
#define YD   64
#define HID  128

typedef __attribute__((ext_vector_type(16))) _Float16 v16h;
typedef __attribute__((ext_vector_type(8)))  float    v8f;
typedef __attribute__((ext_vector_type(2)))  float    v2f;
typedef int v4ig __attribute__((vector_size(16)));   // matches builtin's int4

#define ASPACE(n) __attribute__((address_space(n)))

// Feature probes (evaluated separately for host/device passes; each pass is
// self-consistent, fallbacks keep both compilable).
#if defined(__has_builtin)
#if __has_builtin(__builtin_amdgcn_wmma_f32_16x16x4_f32)
#define HAVE_WMMA_F32 1
#endif
#if __has_builtin(__builtin_amdgcn_global_load_async_to_lds_b128)
#define HAVE_ASYNC_LDS 1
#endif
#endif

// 16-byte global -> LDS copy; async (ASYNCcnt, global_load_async_to_lds_b128)
// when the gfx1250 builtin exists.  Probed signature (from compiler
// diagnostics): (AS1 int4* src, AS3 int4* dst, int offset, int cpol).
// Generic->AS1: identity 64-bit VA.  Generic LDS VA low 32 bits are the LDS
// offset per the aperture scheme, so truncation is the correct AS3 value.
__device__ __forceinline__ void copy16_g2l(const float* g, float* l) {
#if defined(HAVE_ASYNC_LDS) && defined(__AMDGCN__)
    __builtin_amdgcn_global_load_async_to_lds_b128(
        (ASPACE(1) v4ig*)(unsigned long long)g,
        (ASPACE(3) v4ig*)(unsigned int)(unsigned long long)l,
        0, 0);
#else
    *(float4*)l = *(const float4*)g;
#endif
}

__device__ __forceinline__ void wait_async_copies() {
#if defined(HAVE_ASYNC_LDS) && defined(__AMDGCN__)
#if __has_builtin(__builtin_amdgcn_s_wait_asynccnt)
    __builtin_amdgcn_s_wait_asynccnt(0);
#else
    asm volatile("s_wait_asynccnt 0x0" ::: "memory");
#endif
#endif
}

// K index held by (lane, half h) for the 16-bit A/B WMMA fragment (16x32, MxK).
// ISA 7.12.2: lanes 0-15 hold K {0..7, 16..23}, lanes 16-31 hold K {8..15, 24..31}.
__device__ __forceinline__ int wmma_k16(int lane, int h) {
    int kb = (lane & 16) ? 8 : 0;
    int v = h >> 1, p = h & 1;
    return (v < 4) ? (kb + 2 * v + p) : (16 + kb + 2 * (v - 4) + p);
}

// ---------------------------------------------------------------------------
// Pass 1: Ax = x @ W1[:64]  (1024x128),  Ayb = y @ W1[64:] + b1  (1024x128)
// One wave32 per 16x16 output tile.  Preferred path: exact f32 WMMA
// (v_wmma_f32_16x16x4_f32, K=64 as 16 chained steps); fallback: f16 WMMA.
// 1024 tiles total (2 matrices * 64 Mtiles * 8 Ntiles).
// ---------------------------------------------------------------------------
__global__ __launch_bounds__(128)
void k_gemm(const float* __restrict__ x, const float* __restrict__ y,
            const float* __restrict__ W1, const float* __restrict__ b1,
            float* __restrict__ Ax, float* __restrict__ Ayb)
{
    const int wave = blockIdx.x * (blockDim.x >> 5) + (threadIdx.x >> 5);
    const int lane = threadIdx.x & 31;

    const int which = wave >> 9;          // 0 -> x side, 1 -> y side
    const int mt = (wave >> 3) & 63;      // M tile (samples)
    const int nt = wave & 7;              // N tile (hidden)
    const int m0 = mt * 16, n0 = nt * 16;

    const float* __restrict__ S = which ? y : x;
    const int wrow0 = which ? XD : 0;     // row offset into W1

    const int arow = lane & 15;           // A row / B column within tile

    v8f c = {};
#if defined(HAVE_WMMA_F32) && defined(__AMDGCN__)
    // 32-bit A 16x4 layout: lanes 0-15 hold K=0(V0),1(V1); lanes 16-31 K=2,3.
    const int kb2 = (lane & 16) ? 2 : 0;
    #pragma unroll
    for (int kc = 0; kc < 16; ++kc) {     // K = 64 in chunks of 4
        v2f a, b;
        #pragma unroll
        for (int p = 0; p < 2; ++p) {
            const int k = kc * 4 + kb2 + p;
            a[p] = S[(m0 + arow) * XD + k];
            b[p] = W1[(wrow0 + k) * HID + n0 + arow];
        }
        c = __builtin_amdgcn_wmma_f32_16x16x4_f32(
                false, a, false, b, (short)0, c, false, false);
    }
#else
    #pragma unroll
    for (int kc = 0; kc < 2; ++kc) {      // K = 64 in chunks of 32 (f16)
        const int kb = kc * 32;
        v16h a, b;
        #pragma unroll
        for (int h = 0; h < 16; ++h) {
            const int k = wmma_k16(lane, h);
            a[h] = (_Float16)S[(m0 + arow) * XD + kb + k];
            b[h] = (_Float16)W1[(wrow0 + kb + k) * HID + n0 + arow];
        }
        c = __builtin_amdgcn_wmma_f32_16x16x32_f16(
                false, a, false, b, (short)0, c, false, false);
    }
#endif

    float* __restrict__ O = which ? Ayb : Ax;
    const float bias = which ? b1[n0 + (lane & 15)] : 0.0f;
    const int mh = (lane >> 4) * 8;       // C/D layout: rows r (+8 for hi half)
    #pragma unroll
    for (int r = 0; r < 8; ++r)
        O[(m0 + mh + r) * HID + n0 + (lane & 15)] = c[r] + bias;
}

// ---------------------------------------------------------------------------
// log_scores[i] = y[i] . Wb + bb
// ---------------------------------------------------------------------------
__global__ __launch_bounds__(256)
void k_scores(const float* __restrict__ y, const float* __restrict__ Wb,
              const float* __restrict__ bbp, float* __restrict__ ls)
{
    const int i = blockIdx.x * blockDim.x + threadIdx.x;
    if (i >= NS) return;
    float s = bbp[0];
    #pragma unroll 8
    for (int k = 0; k < YD; ++k) s = fmaf(y[i * YD + k], Wb[k], s);
    ls[i] = s;
}

// ---------------------------------------------------------------------------
// Pass 2: pairwise relu-MLP.  Block = 16 Ay rows (i) x 64 Ax rows (j).
// Tiles staged via async global->LDS (ASYNCcnt) when available.
// Emits per-block partial sums of exp(T1 - ls_j) and of diag(T1) (== T0).
// ---------------------------------------------------------------------------
__global__ __launch_bounds__(256)
void k_pair(const float* __restrict__ Ax, const float* __restrict__ Ayb,
            const float* __restrict__ W2, const float* __restrict__ b2p,
            const float* __restrict__ ls,
            float* __restrict__ pExp, float* __restrict__ pDiag)
{
    __shared__ float AxS[64][HID + 4];   // +4 pad: row stride 132 -> conflict-free
    __shared__ float AyS[16][HID + 4];
    __shared__ float w2s[HID];
    __shared__ float lsS[64];
    __shared__ float red[256];

    const int t = threadIdx.x;
    const int jbase = blockIdx.x * 64;
    const int ibase = blockIdx.y * 16;

    // gfx1250 prefetch path (global_prefetch_b8); data is L2-resident anyway.
    if (t < 64) __builtin_prefetch(&Ax[(jbase + t) * HID], 0, 1);

    // Stage tiles in LDS (16B granularity, coalesced; async when supported).
    for (int idx = t; idx < 64 * (HID / 4); idx += 256) {       // 2048 x 16B
        const int r = idx >> 5, cq = (idx & 31) << 2;
        copy16_g2l(&Ax[(jbase + r) * HID + cq], &AxS[r][cq]);
    }
    for (int idx = t; idx < 16 * (HID / 4); idx += 256) {       // 512 x 16B
        const int r = idx >> 5, cq = (idx & 31) << 2;
        copy16_g2l(&Ayb[(ibase + r) * HID + cq], &AyS[r][cq]);
    }
    if (t < HID) w2s[t] = W2[t];
    if (t < 64)  lsS[t] = ls[jbase + t];
    wait_async_copies();
    __syncthreads();

    const int i  = t & 15;        // Ay row (uniform per half-wave j group)
    const int jg = t >> 4;        // 16 groups of 4 j each
    const float b2 = b2p[0];

    float acc[4] = {0.f, 0.f, 0.f, 0.f};
    for (int h = 0; h < HID; h += 4) {
        const float4 ay = *(const float4*)&AyS[i][h];
        const float4 w  = *(const float4*)&w2s[h];
        #pragma unroll
        for (int q = 0; q < 4; ++q) {
            const float4 ax = *(const float4*)&AxS[jg * 4 + q][h];
            acc[q] += fmaxf(ax.x + ay.x, 0.f) * w.x
                    + fmaxf(ax.y + ay.y, 0.f) * w.y
                    + fmaxf(ax.z + ay.z, 0.f) * w.z
                    + fmaxf(ax.w + ay.w, 0.f) * w.w;
        }
    }

    float lexp = 0.f, ldiag = 0.f;
    #pragma unroll
    for (int q = 0; q < 4; ++q) {
        const int j = jg * 4 + q;
        const float t1 = acc[q] + b2;
        lexp += __expf(t1 - lsS[j]);
        if (jbase + j == ibase + i) ldiag = t1;   // T0 contribution (diag of T1)
    }

    // Deterministic block tree reductions.
    red[t] = lexp; __syncthreads();
    #pragma unroll
    for (int s = 128; s > 0; s >>= 1) {
        if (t < s) red[t] += red[t + s];
        __syncthreads();
    }
    if (t == 0) pExp[blockIdx.y * gridDim.x + blockIdx.x] = red[0];
    __syncthreads();

    red[t] = ldiag; __syncthreads();
    #pragma unroll
    for (int s = 128; s > 0; s >>= 1) {
        if (t < s) red[t] += red[t + s];
        __syncthreads();
    }
    if (t == 0) pDiag[blockIdx.y * gridDim.x + blockIdx.x] = red[0];
}

// ---------------------------------------------------------------------------
// Final: lower = 1 + sum(diag)/N - sum(ls)/N - sum(exp)/N^2
// ---------------------------------------------------------------------------
__global__ __launch_bounds__(256)
void k_final(const float* __restrict__ ls, const float* __restrict__ pExp,
             const float* __restrict__ pDiag, float* __restrict__ out)
{
    __shared__ float rE[256], rD[256], rL[256];
    const int t = threadIdx.x;
    float se = 0.f, sd = 0.f, sl = 0.f;
    for (int k = t; k < 1024; k += 256) {
        se += pExp[k];
        sd += pDiag[k];
        sl += ls[k];
    }
    rE[t] = se; rD[t] = sd; rL[t] = sl; __syncthreads();
    #pragma unroll
    for (int s = 128; s > 0; s >>= 1) {
        if (t < s) { rE[t] += rE[t + s]; rD[t] += rD[t + s]; rL[t] += rL[t + s]; }
        __syncthreads();
    }
    if (t == 0) {
        const float invN = 1.0f / (float)NS;
        out[0] = 1.0f + rD[0] * invN - rL[0] * invN - rE[0] * invN * invN;
    }
}

extern "C" void kernel_launch(void* const* d_in, const int* in_sizes, int n_in,
                              void* d_out, int out_size, void* d_ws, size_t ws_size,
                              hipStream_t stream)
{
    (void)in_sizes; (void)n_in; (void)out_size; (void)ws_size;

    const float* x  = (const float*)d_in[0];   // (1024, 64)
    const float* y  = (const float*)d_in[1];   // (1024, 64)
    const float* W1 = (const float*)d_in[2];   // (128, 128)
    const float* b1 = (const float*)d_in[3];   // (128,)
    const float* W2 = (const float*)d_in[4];   // (128,)
    const float* b2 = (const float*)d_in[5];   // scalar
    const float* Wb = (const float*)d_in[6];   // (64,)
    const float* bb = (const float*)d_in[7];   // scalar
    float* out = (float*)d_out;

    float* wsf  = (float*)d_ws;
    float* Ax   = wsf;                 // 1024*128
    float* Ayb  = Ax  + NS * HID;      // 1024*128
    float* ls   = Ayb + NS * HID;      // 1024
    float* pE   = ls  + NS;            // 1024 per-block exp partials
    float* pD   = pE  + NS;            // 1024 per-block diag partials

    // 1024 waves = 2 matrices * 64 Mtiles * 8 Ntiles; 4 waves/block.
    k_gemm  <<<256, 128, 0, stream>>>(x, y, W1, b1, Ax, Ayb);
    k_scores<<<4,   256, 0, stream>>>(y, Wb, bb, ls);
    k_pair  <<<dim3(16, 64), 256, 0, stream>>>(Ax, Ayb, W2, b2, ls, pE, pD);
    k_final <<<1,   256, 0, stream>>>(ls, pE, pD, out);
}
